// ComplexPtreeLayer_47236050321808
// MI455X (gfx1250) — matrix operand order repaired and checked by reading, not compiled, not run on hardware
//
#include <hip/hip_runtime.h>
#include <hip/hip_bf16.h>

// ---------------------------------------------------------------------------
// Problem constants (from reference): K=4, H=512, B=4, T=1024, D=3, N0=65536.
// Exact algebraic collapse:
//   level(data)[g] = Weff @ (sum of 4 children) + beff
//   Weff = Wzf @ (sum_k Wz[k]),  beff = 4*(Wzf @ sum_k bz[k] + bzf)
// Pipeline: precompute Weff/beff once, then 3x (pool4 -> GEMM 512x512).
// GEMM: bf16x3 split (hi*hi + hi*lo + lo*hi) on v_wmma_f32_16x16x32_bf16.
// A strip staged in LDS via async global->LDS (ASYNCcnt) and reused by all
// 16 waves of the block; B (Weff) streamed from L2.
// ---------------------------------------------------------------------------

#define H 512
#define HMASK 511
#define HSHIFT 9

typedef __bf16 bf16_t;
typedef bf16_t v16bf __attribute__((ext_vector_type(16)));
typedef bf16_t v8bf  __attribute__((ext_vector_type(8)));
typedef float  v8f   __attribute__((ext_vector_type(8)));
typedef int    v4i   __attribute__((ext_vector_type(4)));

// LDS A strip: [hi|lo][32 rows][512 + 8 pad] bf16  -> 66,560 bytes
#define LDS_ROW 520
#define LDS_MAT (32 * LDS_ROW)

// ---------------------------------------------------------------------------
// Async global->LDS copy of one 16B chunk (falls back to sync load+ds_store).
// Builtin signature (per hipcc diagnostic): (AS1 v4i*, AS3 v4i*, imm, imm).
// ---------------------------------------------------------------------------
#if __has_builtin(__builtin_amdgcn_global_load_async_to_lds_b128)
#define HAVE_ASYNC_LDS 1
typedef __attribute__((address_space(1))) v4i* g_v4i_ptr;
typedef __attribute__((address_space(3))) v4i* l_v4i_ptr;
static __device__ inline void copy16_g2l(const bf16_t* g, bf16_t* l) {
  __builtin_amdgcn_global_load_async_to_lds_b128(
      (g_v4i_ptr)(v4i*)(bf16_t*)g, (l_v4i_ptr)(v4i*)l, 0, 0);
}
static __device__ inline void wait_async_copies() {
#if __has_builtin(__builtin_amdgcn_s_wait_asynccnt)
  __builtin_amdgcn_s_wait_asynccnt(0);
#else
  asm volatile("s_wait_asynccnt 0x0" ::: "memory");
#endif
}
#else
#define HAVE_ASYNC_LDS 0
static __device__ inline void copy16_g2l(const bf16_t* g, bf16_t* l) {
  *(v8bf*)l = *(const v8bf*)g;     // global_load_b128 + ds_store_b128
}
static __device__ inline void wait_async_copies() {}
#endif

// ---------------------------------------------------------------------------
// Wsum[m,h] = sum_k Wz[k,m,h]   (Wz: [4,512,512])
// ---------------------------------------------------------------------------
__global__ __launch_bounds__(256) void wsum_kernel(const float* __restrict__ Wz,
                                                   float* __restrict__ Wsum) {
  int idx = blockIdx.x * 256 + threadIdx.x;         // 0 .. 512*512-1
  const int HH = H * H;
  Wsum[idx] = Wz[idx] + Wz[HH + idx] + Wz[2 * HH + idx] + Wz[3 * HH + idx];
}

// ---------------------------------------------------------------------------
// Weff[j,h] = sum_m Wzf[j,m] * Wsum[m,h]; write bf16 hi/lo split.
// ---------------------------------------------------------------------------
__global__ __launch_bounds__(256) void weff_kernel(const float* __restrict__ Wzf,
                                                   const float* __restrict__ Wsum,
                                                   bf16_t* __restrict__ Whi,
                                                   bf16_t* __restrict__ Wlo) {
  int idx = blockIdx.x * 256 + threadIdx.x;         // 0 .. 512*512-1
  int h = idx & HMASK;
  int j = idx >> HSHIFT;
  const float* wzf_row = Wzf + (size_t)j * H;
  float acc = 0.f;
  #pragma unroll 8
  for (int m = 0; m < H; ++m) {
    acc = __builtin_fmaf(wzf_row[m], Wsum[(size_t)m * H + h], acc);
  }
  bf16_t hi = (bf16_t)acc;
  float  res = acc - (float)hi;
  Whi[idx] = hi;
  Wlo[idx] = (bf16_t)res;
}

// ---------------------------------------------------------------------------
// beff[j] = 4 * ( sum_m Wzf[j,m] * bsum[m] + bzf[j] ),  bsum[m] = sum_k bz[k,m]
// ---------------------------------------------------------------------------
__global__ __launch_bounds__(512) void beff_kernel(const float* __restrict__ Wzf,
                                                   const float* __restrict__ bzf,
                                                   const float* __restrict__ bz,
                                                   float* __restrict__ beff) {
  __shared__ float bs[H];
  int t = threadIdx.x;                               // 512 threads
  bs[t] = bz[t] + bz[H + t] + bz[2 * H + t] + bz[3 * H + t];
  __syncthreads();
  const float* wzf_row = Wzf + (size_t)t * H;
  float acc = 0.f;
  #pragma unroll 8
  for (int m = 0; m < H; ++m) acc = __builtin_fmaf(wzf_row[m], bs[m], acc);
  beff[t] = 4.0f * (acc + bzf[t]);
}

// ---------------------------------------------------------------------------
// Pool groups of 4 rows (optional index gather on level 1), split to bf16 hi/lo.
// ---------------------------------------------------------------------------
__global__ __launch_bounds__(256) void pool4_split_kernel(const float* __restrict__ in,
                                                          const int* __restrict__ gidx,
                                                          int Nout,
                                                          bf16_t* __restrict__ shi,
                                                          bf16_t* __restrict__ slo) {
  int idx = blockIdx.x * 256 + threadIdx.x;
  if (idx >= Nout * H) return;
  int h = idx & HMASK;
  int g = idx >> HSHIFT;
  float s = 0.f;
  #pragma unroll
  for (int q = 0; q < 4; ++q) {
    int r = 4 * g + q;
    if (gidx) r = gidx[r];
    s += in[(size_t)r * H + h];
  }
  bf16_t hi = (bf16_t)s;
  float  res = s - (float)hi;
  shi[idx] = hi;
  slo[idx] = (bf16_t)res;
}

// ---------------------------------------------------------------------------
// WMMA fragment loads (16x16x32 bf16, wave32; this code's own k order, shared by A and B).
// A (16x32, MxK): lane = 16*hi + m ; holds K = 16*hi + 0..15 (contiguous)
// B (32x16, KxN): lane = 16*hi + n ; holds K = 16*hi + 0..15 (contiguous)
// ---------------------------------------------------------------------------
static __device__ inline v16bf load_fragA_lds(const bf16_t* row, int kb, int hi) {
  union { v16bf v; v8bf h[2]; } u;
  u.h[0] = *(const v8bf*)(row + kb + 16 * hi);          // ds_load_b128
  u.h[1] = *(const v8bf*)(row + kb + 16 * hi + 8);      // ds_load_b128
  return u.v;
}
static __device__ inline v16bf load_fragB(const bf16_t* row, int kb, int hi) {
  return *(const v16bf*)(row + kb + 16 * hi);           // global_load x2
}

#define WMMA_BF16(C, A, B)                                                      \
  C = __builtin_amdgcn_wmma_f32_16x16x32_bf16(false, (A), false, (B),           \
                                              (short)0, (C), false, false)

// ---------------------------------------------------------------------------
// out[M,512] = S[M,512] @ Weff^T + beff, bf16x3 split accumulation in f32.
// One block = 512 threads = 16 waves = one 32(M) x 512(N) output strip.
// A strip (hi+lo) staged once in LDS via async copies; each wave owns a
// 32x32 tile (2x2 WMMA subtiles) and reads A from LDS, B from global/L2.
// ---------------------------------------------------------------------------
__global__ __launch_bounds__(512) void gemm_bf16x3_kernel(const bf16_t* __restrict__ Shi,
                                                          const bf16_t* __restrict__ Slo,
                                                          const bf16_t* __restrict__ Whi,
                                                          const bf16_t* __restrict__ Wlo,
                                                          const float* __restrict__ beff,
                                                          float* __restrict__ out) {
  __shared__ alignas(16) bf16_t ldsA[2 * LDS_MAT];

  const int tid  = threadIdx.x;
  const int m0   = blockIdx.x * 32;

  // ---- Stage A strip (32 rows x 512 k, hi then lo) into LDS: 16B chunks ----
  // 2 matrices x 32 rows x 64 chunks = 4096 chunks; 8 per thread, coalesced.
  #pragma unroll
  for (int mat = 0; mat < 2; ++mat) {
    const bf16_t* src = mat ? Slo : Shi;
    #pragma unroll
    for (int it = 0; it < 4; ++it) {
      int chunk = it * 512 + tid;                   // 0..2047
      int r = chunk >> 6;                           // row 0..31
      int c = chunk & 63;                           // 16B chunk in row
      const bf16_t* g = src + (((size_t)(m0 + r)) << HSHIFT) + c * 8;
      bf16_t* l = ldsA + mat * LDS_MAT + r * LDS_ROW + c * 8;
      copy16_g2l(g, l);
    }
  }
  wait_async_copies();
  __syncthreads();

  // ---- Per-wave 32x32 tile ----
  const int wave = tid >> 5;                        // 0..15 -> n strip
  const int lane = tid & 31;
  const int n0 = wave * 32;
  const int lr = lane & 15;
  const int hi = lane >> 4;

  const bf16_t* a0h = ldsA + (lr) * LDS_ROW;                 // rows 0..15 (hi)
  const bf16_t* a1h = ldsA + (16 + lr) * LDS_ROW;            // rows 16..31 (hi)
  const bf16_t* a0l = ldsA + LDS_MAT + (lr) * LDS_ROW;       // rows 0..15 (lo)
  const bf16_t* a1l = ldsA + LDS_MAT + (16 + lr) * LDS_ROW;  // rows 16..31 (lo)

  const bf16_t* b0h = Whi + (size_t)(n0 + lr) * H;
  const bf16_t* b1h = Whi + (size_t)(n0 + 16 + lr) * H;
  const bf16_t* b0l = Wlo + (size_t)(n0 + lr) * H;
  const bf16_t* b1l = Wlo + (size_t)(n0 + 16 + lr) * H;

  float bias0 = beff[n0 + lr];
  float bias1 = beff[n0 + 16 + lr];
  v8f c00, c01, c10, c11;
  #pragma unroll
  for (int r = 0; r < 8; ++r) { c00[r] = bias0; c01[r] = bias1; c10[r] = bias0; c11[r] = bias1; }

  for (int kb = 0; kb < H; kb += 32) {
    v16bf bh0 = load_fragB(b0h, kb, hi);
    v16bf bh1 = load_fragB(b1h, kb, hi);
    v16bf bl0 = load_fragB(b0l, kb, hi);
    v16bf bl1 = load_fragB(b1l, kb, hi);
    v16bf ah0 = load_fragA_lds(a0h, kb, hi);
    v16bf ah1 = load_fragA_lds(a1h, kb, hi);
    v16bf al0 = load_fragA_lds(a0l, kb, hi);
    v16bf al1 = load_fragA_lds(a1l, kb, hi);

    WMMA_BF16(c00, ah0, bh0);  WMMA_BF16(c01, ah0, bh1);
    WMMA_BF16(c10, ah1, bh0);  WMMA_BF16(c11, ah1, bh1);

    WMMA_BF16(c00, ah0, bl0);  WMMA_BF16(c01, ah0, bl1);
    WMMA_BF16(c10, ah1, bl0);  WMMA_BF16(c11, ah1, bl1);

    WMMA_BF16(c00, al0, bh0);  WMMA_BF16(c01, al0, bh1);
    WMMA_BF16(c10, al1, bh0);  WMMA_BF16(c11, al1, bh1);
  }

  // D layout (16x16 f32): lane(0-15)=col n, vgpr r = row r ; lane(16-31): rows 8-15.
  int col0 = n0 + lr;
  int col1 = n0 + 16 + lr;
  #pragma unroll
  for (int r = 0; r < 8; ++r) {
    int row0 = m0 + 8 * hi + r;
    int row1 = row0 + 16;
    out[(size_t)row0 * H + col0] = c00[r];
    out[(size_t)row0 * H + col1] = c01[r];
    out[(size_t)row1 * H + col0] = c10[r];
    out[(size_t)row1 * H + col1] = c11[r];
  }
}

// ---------------------------------------------------------------------------
// Launch
// ---------------------------------------------------------------------------
extern "C" void kernel_launch(void* const* d_in, const int* in_sizes, int n_in,
                              void* d_out, int out_size, void* d_ws, size_t ws_size,
                              hipStream_t stream) {
  const float* x    = (const float*)d_in[0];   // [65536, 512]
  const float* Wz   = (const float*)d_in[1];   // [4, 512, 512]
  const float* bz   = (const float*)d_in[2];   // [4, 512]
  const float* Wzf  = (const float*)d_in[3];   // [512, 512]
  const float* bzf  = (const float*)d_in[4];   // [512]
  const int*   nidx = (const int*)d_in[5];     // [65536]
  float* out = (float*)d_out;                  // [1024, 512]

  // Workspace layout (bytes), 256-aligned.
  char* ws = (char*)d_ws;
  float*  Wsum = (float*)(ws + 0);                         //  1,048,576
  bf16_t* Whi  = (bf16_t*)(ws + 1048576);                  //    524,288
  bf16_t* Wlo  = (bf16_t*)(ws + 1572864);                  //    524,288
  float*  beff = (float*)(ws + 2097152);                   //      2,048
  bf16_t* Shi  = (bf16_t*)(ws + 2099200);                  // 16,777,216
  bf16_t* Slo  = (bf16_t*)(ws + 18876416);                 // 16,777,216
  float*  dbuf = (float*)(ws + 35653632);                  // 33,554,432  (f32 inter-level)
  (void)in_sizes; (void)n_in; (void)out_size; (void)ws_size;

  // --- Precompute effective weight/bias ---
  wsum_kernel<<<(H * H) / 256, 256, 0, stream>>>(Wz, Wsum);
  weff_kernel<<<(H * H) / 256, 256, 0, stream>>>(Wzf, Wsum, Whi, Wlo);
  beff_kernel<<<1, 512, 0, stream>>>(Wzf, bzf, bz, beff);

  // --- Level 1: pool 65536 -> 16384 (with node_idx gather), GEMM ---
  pool4_split_kernel<<<(16384 * H) / 256, 256, 0, stream>>>(x, nidx, 16384, Shi, Slo);
  gemm_bf16x3_kernel<<<16384 / 32, 512, 0, stream>>>(Shi, Slo, Whi, Wlo, beff, dbuf);

  // --- Level 2: pool 16384 -> 4096, GEMM (dbuf overwritten after being pooled) ---
  pool4_split_kernel<<<(4096 * H) / 256, 256, 0, stream>>>(dbuf, nullptr, 4096, Shi, Slo);
  gemm_bf16x3_kernel<<<4096 / 32, 512, 0, stream>>>(Shi, Slo, Whi, Wlo, beff, dbuf);

  // --- Level 3: pool 4096 -> 1024, GEMM -> d_out ---
  pool4_split_kernel<<<(1024 * H) / 256, 256, 0, stream>>>(dbuf, nullptr, 1024, Shi, Slo);
  gemm_bf16x3_kernel<<<1024 / 32, 512, 0, stream>>>(Shi, Slo, Whi, Wlo, beff, out);
}
